// MLBAttention_82738249990598
// MI455X (gfx1250) — matrix operand order repaired and checked by reading, not compiled, not run on hardware
//
#include <hip/hip_runtime.h>
#include <hip/hip_bf16.h>
#include <cstdint>

#define B_ 32
#define N_ 2048
#define D_ 1024
#define C_ 1024
#define H_ 1024
#define NEGBIG (-1e30f)

typedef __bf16 bf16_t;
typedef __attribute__((ext_vector_type(16))) __bf16 v16bf;
typedef __attribute__((ext_vector_type(8)))  __bf16 v8bf;
typedef __attribute__((ext_vector_type(8)))  float  v8f;

// ---------------------------------------------------------------------------
// init: zero score accumulator + output (both are atomically accumulated)
// ---------------------------------------------------------------------------
__global__ __launch_bounds__(256) void k_init(float* __restrict__ scores,
                                              float* __restrict__ out) {
    int i = blockIdx.x * 256 + threadIdx.x;
    if (i < B_ * N_) scores[i] = 0.f;
    if (i < B_ * D_) out[i] = 0.f;
}

// ---------------------------------------------------------------------------
// convert W_ih [H,D] f32 -> bf16 once (2 MB, L2 resident afterwards)
// ---------------------------------------------------------------------------
__global__ __launch_bounds__(256) void k_wcvt(const float* __restrict__ W,
                                              bf16_t* __restrict__ Wb) {
    int i = (blockIdx.x * 256 + threadIdx.x) * 8;
    float4 f0 = *(const float4*)(W + i);
    float4 f1 = *(const float4*)(W + i + 4);
    v8bf o;
    o[0] = (bf16_t)f0.x; o[1] = (bf16_t)f0.y; o[2] = (bf16_t)f0.z; o[3] = (bf16_t)f0.w;
    o[4] = (bf16_t)f1.x; o[5] = (bf16_t)f1.y; o[6] = (bf16_t)f1.z; o[7] = (bf16_t)f1.w;
    *(v8bf*)(Wb + i) = o;
}

// ---------------------------------------------------------------------------
// cw[b,h] = tanh(context[b,:] . W_ch[h,:]) * w_score[h]   (tiny: 67 MFLOP)
// ---------------------------------------------------------------------------
__global__ __launch_bounds__(256) void k_cw(const float* __restrict__ context,
                                            const float* __restrict__ W_ch,
                                            const float* __restrict__ w_score,
                                            float* __restrict__ cw) {
    int idx = blockIdx.x * 256 + threadIdx.x;   // 0 .. B_*H_-1
    int b = idx >> 10;
    int h = idx & (H_ - 1);
    const float* cr = context + b * C_;
    const float* wr = W_ch + (size_t)h * C_;
    float s = 0.f;
    for (int c = 0; c < C_; c += 4) {
        float4 cv = *(const float4*)(cr + c);
        float4 wv = *(const float4*)(wr + c);
        s += cv.x * wv.x + cv.y * wv.y + cv.z * wv.z + cv.w * wv.w;
    }
    cw[idx] = tanhf(s) * w_score[h];
}

// ---------------------------------------------------------------------------
// Main fused kernel: scores[b,n] = sum_h tanh((inputs @ W_ih^T)[b,n,h]) * cw[b,h]
// WG = 256 threads (8 waves), owns 64 rows of batch b.
// Wave w covers 2 h-chunks per g pass (8 waves * 2 = 16 chunks, H/16=64 -> g<4).
// bf16 WMMA 16x16x32, f32 accumulate; M-block x4, H-block x2 -> 8 WMMA / K-step.
// A K-slab (64 rows x 32 K) staged f32->bf16 in double-buffered LDS (8KB).
// ---------------------------------------------------------------------------
__global__ __launch_bounds__(256) void k_scores(const float* __restrict__ inputs,
                                                const bf16_t* __restrict__ Wb,
                                                const float* __restrict__ cw,
                                                float* __restrict__ scores) {
    __shared__ bf16_t atile[2][64 * 32];        // 8 KB double buffer

    const int b    = blockIdx.y;
    const int n0   = blockIdx.x * 64;
    const int tid  = threadIdx.x;
    const int wave = tid >> 5;
    const int lane = tid & 31;
    const int lane16 = lane & 15;
    const int hi     = lane >> 4;               // 0 | 1 (lane half)

    const float* __restrict__ inpB = inputs + ((size_t)b * N_ + n0) * D_;

    // staging map: thread converts 8 consecutive f32 of the 64x32 slab
    const int srow = tid >> 2;                  // 0..63
    const int scol = (tid & 3) << 3;            // 0,8,16,24

    for (int g = 0; g < 4; ++g) {
        // this wave's two 16-wide h chunks
        const int   h0   = (g * 16 + wave * 2) * 16;
        const int   h1   = h0 + 16;
        const float cwv0 = cw[b * H_ + h0 + lane16];
        const float cwv1 = cw[b * H_ + h1 + lane16];

        v8f acc0[4], acc1[4];
#pragma unroll
        for (int m = 0; m < 4; ++m) {
            acc0[m] = (v8f){0, 0, 0, 0, 0, 0, 0, 0};
            acc1[m] = (v8f){0, 0, 0, 0, 0, 0, 0, 0};
        }

        // prologue: stage K-step 0 into buffer 0
        {
            const float* src = inpB + (size_t)srow * D_ + scol;
            float4 f0 = *(const float4*)src;
            float4 f1 = *(const float4*)(src + 4);
            v8bf o;
            o[0] = (bf16_t)f0.x; o[1] = (bf16_t)f0.y; o[2] = (bf16_t)f0.z; o[3] = (bf16_t)f0.w;
            o[4] = (bf16_t)f1.x; o[5] = (bf16_t)f1.y; o[6] = (bf16_t)f1.z; o[7] = (bf16_t)f1.w;
            *(v8bf*)&atile[0][srow * 32 + scol] = o;
        }
        __syncthreads();

        for (int k = 0; k < 32; ++k) {
            // stage next K-slab into the other buffer (overlaps with compute)
            if (k + 1 < 32) {
                const float* src = inpB + (size_t)srow * D_ + (k + 1) * 32 + scol;
                float4 f0 = *(const float4*)src;
                float4 f1 = *(const float4*)(src + 4);
                v8bf o;
                o[0] = (bf16_t)f0.x; o[1] = (bf16_t)f0.y; o[2] = (bf16_t)f0.z; o[3] = (bf16_t)f0.w;
                o[4] = (bf16_t)f1.x; o[5] = (bf16_t)f1.y; o[6] = (bf16_t)f1.z; o[7] = (bf16_t)f1.w;
                *(v8bf*)&atile[(k + 1) & 1][srow * 32 + scol] = o;
            }

            // B fragments: 32(K=d) x 16(N=h); lane holds col h+lane16,
            // K runs {hi*8..hi*8+7, 16+hi*8..16+hi*8+7} (matches A layout).
            const v8bf* wp0 =
                (const v8bf*)(Wb + (size_t)(h0 + lane16) * D_ + k * 32 + hi * 8);
            const v8bf* wp1 =
                (const v8bf*)(Wb + (size_t)(h1 + lane16) * D_ + k * 32 + hi * 8);
            v8bf b00 = wp0[0], b01 = wp0[2];
            v8bf b10 = wp1[0], b11 = wp1[2];
            v16bf bfrag0 = __builtin_shufflevector(b00, b01, 0, 1, 2, 3, 4, 5, 6, 7,
                                                   8, 9, 10, 11, 12, 13, 14, 15);
            v16bf bfrag1 = __builtin_shufflevector(b10, b11, 0, 1, 2, 3, 4, 5, 6, 7,
                                                   8, 9, 10, 11, 12, 13, 14, 15);

            // A fragments first (one dscnt wait), then a burst of 8 WMMAs.
            // rows m*16+lane16; lane<16 K runs {0..7,16..23},
            // lane>=16 K runs {8..15,24..31}  (ISA 16-bit 16x32 layout)
            const bf16_t* abase = &atile[k & 1][0];
            v16bf afrag[4];
#pragma unroll
            for (int m = 0; m < 4; ++m) {
                const bf16_t* ap = abase + (m * 16 + lane16) * 32 + hi * 8;
                v8bf a0 = *(const v8bf*)ap;          // K = k0 .. k0+7
                v8bf a1 = *(const v8bf*)(ap + 16);   // K = k0+16 .. k0+23
                afrag[m] = __builtin_shufflevector(a0, a1, 0, 1, 2, 3, 4, 5, 6, 7,
                                                   8, 9, 10, 11, 12, 13, 14, 15);
            }
#pragma unroll
            for (int m = 0; m < 4; ++m) {
                acc0[m] = __builtin_amdgcn_wmma_f32_16x16x32_bf16(
                    false, afrag[m], false, bfrag0, (short)0, acc0[m], false, false);
            }
#pragma unroll
            for (int m = 0; m < 4; ++m) {
                acc1[m] = __builtin_amdgcn_wmma_f32_16x16x32_bf16(
                    false, afrag[m], false, bfrag1, (short)0, acc1[m], false, false);
            }
            __syncthreads();
        }

        // epilogue: tanh, scale by cw, reduce 16 h-columns (lanes) per row,
        // accumulate into scores. acc[m][r]: row = m*16 + hi*8 + r, col = lane16.
#pragma unroll
        for (int m = 0; m < 4; ++m) {
#pragma unroll
            for (int r = 0; r < 8; ++r) {
                float v = tanhf(acc0[m][r]) * cwv0 + tanhf(acc1[m][r]) * cwv1;
                v += __shfl_xor(v, 1, 32);
                v += __shfl_xor(v, 2, 32);
                v += __shfl_xor(v, 4, 32);
                v += __shfl_xor(v, 8, 32);
                if (lane16 == 0) {
                    atomicAdd(&scores[b * N_ + n0 + m * 16 + hi * 8 + r], v);
                }
            }
        }
    }
}

// ---------------------------------------------------------------------------
// masked softmax over N per batch (in-place: scores -> attn)
// ---------------------------------------------------------------------------
__global__ __launch_bounds__(256) void k_softmax(float* __restrict__ scores,
                                                 const unsigned char* __restrict__ mask) {
    __shared__ float buf[N_];   // 8 KB
    __shared__ float red[8];
    const int b = blockIdx.x;
    const int tid = threadIdx.x;

    float mx = NEGBIG;
    for (int n = tid; n < N_; n += 256) {
        float s = scores[b * N_ + n];
        s = mask[b * N_ + n] ? s : NEGBIG;
        buf[n] = s;
        mx = fmaxf(mx, s);
    }
#pragma unroll
    for (int m = 16; m >= 1; m >>= 1) mx = fmaxf(mx, __shfl_xor(mx, m, 32));
    if ((tid & 31) == 0) red[tid >> 5] = mx;
    __syncthreads();
    float mall = red[0];
#pragma unroll
    for (int i = 1; i < 8; ++i) mall = fmaxf(mall, red[i]);

    float sum = 0.f;
    for (int n = tid; n < N_; n += 256) {
        float e = __expf(buf[n] - mall);
        buf[n] = e;
        sum += e;
    }
#pragma unroll
    for (int m = 16; m >= 1; m >>= 1) sum += __shfl_xor(sum, m, 32);
    __syncthreads();
    if ((tid & 31) == 0) red[tid >> 5] = sum;
    __syncthreads();
    float sall = 0.f;
#pragma unroll
    for (int i = 0; i < 8; ++i) sall += red[i];
    float inv = 1.f / sall;

    for (int n = tid; n < N_; n += 256) scores[b * N_ + n] = buf[n] * inv;
}

// ---------------------------------------------------------------------------
// out[b,d] = sum_n attn[b,n] * inputs[b,n,d]  (N split 8 ways, atomic combine)
// ---------------------------------------------------------------------------
__global__ __launch_bounds__(256) void k_out(const float* __restrict__ inputs,
                                             const float* __restrict__ attn,
                                             float* __restrict__ out) {
    const int b = blockIdx.y;
    const int d = blockIdx.x * 256 + threadIdx.x;
    const int n0 = blockIdx.z * 256;
    const float* ip = inputs + ((size_t)b * N_ + n0) * D_ + d;
    const float* ap = attn + b * N_ + n0;
    float acc = 0.f;
#pragma unroll 4
    for (int n = 0; n < 256; ++n) acc += ap[n] * ip[(size_t)n * D_];
    atomicAdd(&out[b * D_ + d], acc);
}

// ---------------------------------------------------------------------------
extern "C" void kernel_launch(void* const* d_in, const int* in_sizes, int n_in,
                              void* d_out, int out_size, void* d_ws, size_t ws_size,
                              hipStream_t stream) {
    const float* context         = (const float*)d_in[0];  // [B,C]
    const float* inputs          = (const float*)d_in[1];  // [B,N,D]
    const unsigned char* mask    = (const unsigned char*)d_in[2];  // [B,N,1] bool
    const float* W_ih            = (const float*)d_in[3];  // [H,D]
    const float* W_ch            = (const float*)d_in[4];  // [H,C]
    const float* w_score         = (const float*)d_in[5];  // [H]
    float* out                   = (float*)d_out;          // [B,1,D]

    char* ws = (char*)d_ws;
    bf16_t* Wb     = (bf16_t*)ws;                            // H*D bf16  = 2 MB
    float*  cw     = (float*)(ws + (size_t)H_ * D_ * 2);     // B*H f32   = 128 KB
    float*  scores = cw + B_ * H_;                           // B*N f32   = 256 KB

    k_init<<<dim3((B_ * N_ + 255) / 256), 256, 0, stream>>>(scores, out);
    k_wcvt<<<dim3(H_ * D_ / (256 * 8)), 256, 0, stream>>>(W_ih, Wb);
    k_cw<<<dim3(B_ * H_ / 256), 256, 0, stream>>>(context, W_ch, w_score, cw);
    k_scores<<<dim3(N_ / 64, B_), 256, 0, stream>>>(inputs, Wb, cw, scores);
    k_softmax<<<dim3(B_), 256, 0, stream>>>(scores, mask);
    k_out<<<dim3(D_ / 256, B_, 8), 256, 0, stream>>>(inputs, scores, out);
}
